// GMMBaseDistribution_26328149524578
// MI455X (gfx1250) — matrix operand order repaired and checked by reading, not compiled, not run on hardware
//
#include <hip/hip_runtime.h>
#include <math.h>

// ---------------------------------------------------------------------------
// GMM marginal log-prob as  [z | z^2] (16384x512) x W (512x64)  + bias,
// then row-wise logsumexp.  GEMM runs on v_wmma_f32_16x16x32_bf16 with a
// hi/lo bf16 split of both operands (3 products) for ~f32 accuracy.
//
//   B = 16384 rows, F = 256 features, C = 64 classes.
//   w1[c,f] = -2*mu*iv,  w0[c,f] = iv = exp(-2*ls)
//   bias[c] = F*log(2pi) + sum_f (2*ls + mu^2*iv)
//   lp[b,c] = -0.5*( S[b,c] + bias[c] ),  S = z.w1 + z^2.w0
//   out[b]  = -log(C) + logsumexp_c lp[b,c]
//
// ILP: per class-tile we keep TWO accumulator chains (z-term and z^2-term)
// and interleave their WMMAs, so adjacent v_wmma ops are independent
// (dependency depth 3 instead of 6 on the XDL pipe).
// ---------------------------------------------------------------------------

typedef __attribute__((ext_vector_type(16))) __bf16 v16bf;
typedef __attribute__((ext_vector_type(8)))  float  v8f;

#define GMM_F      256
#define GMM_C      64
#define LOG2PI     1.83787706640934534f
#define LOG_C      4.15888308335967186f  // log(64)

// Packed-W addressing (matches V_WMMA_F32_16X16X32_BF16 B-matrix layout):
//   tile t = c/16, n = c%16, feature group g = f/32, kl = f%32
//   chunk index: g for w1 (z term), 8+g for w0 (z^2 term)   -> 16 chunks/tile
//   lane = (((kl>>3)&1) ? 16+n : n), element e = (kl&7) | ((kl>>4)<<3)
// Flat bf16 index: ((t*16 + chunk)*32 + lane)*16 + e, total 4*16*32*16 = 32768
#define WPK_ELEMS  32768                // per hi / per lo array

__global__ void gmm_prep_kernel(const float* __restrict__ means,
                                const float* __restrict__ log_stds,
                                __bf16* __restrict__ Whi,
                                __bf16* __restrict__ Wlo,
                                float*  __restrict__ bias)
{
    const int c = blockIdx.x;        // class 0..63
    const int f = threadIdx.x;       // feature 0..255
    const float m  = means[c * GMM_F + f];
    const float ls = log_stds[c * GMM_F + f];
    const float iv = expf(-2.0f * ls);       // 1/sigma^2
    const float w1 = -2.0f * m * iv;         // coeff of z
    const float w0 = iv;                     // coeff of z^2

    const int t    = c >> 4;
    const int n    = c & 15;
    const int g    = f >> 5;
    const int kl   = f & 31;
    const int lane = ((kl >> 3) & 1) ? (16 + n) : n;
    const int e    = (kl & 7) | ((kl >> 4) << 3);
    const int i1   = ((t * 16 + g)     * 32 + lane) * 16 + e;  // w1 slot
    const int i0   = ((t * 16 + 8 + g) * 32 + lane) * 16 + e;  // w0 slot

    __bf16 h;
    h = (__bf16)w1; Whi[i1] = h; Wlo[i1] = (__bf16)(w1 - (float)h);
    h = (__bf16)w0; Whi[i0] = h; Wlo[i0] = (__bf16)(w0 - (float)h);

    // per-class bias reduction
    __shared__ float red[GMM_F];
    red[f] = m * m * iv + 2.0f * ls;
    __syncthreads();
    #pragma unroll
    for (int s = GMM_F / 2; s > 0; s >>= 1) {
        if (f < s) red[f] += red[f + s];
        __syncthreads();
    }
    if (f == 0) bias[c] = (float)GMM_F * LOG2PI + red[0];
}

__launch_bounds__(256)
__global__ void gmm_main_kernel(const float* __restrict__ z,
                                const __bf16* __restrict__ Whi,
                                const __bf16* __restrict__ Wlo,
                                const float* __restrict__ bias,
                                float* __restrict__ out,
                                int nrows)
{
    const int lane = threadIdx.x & 31;
    const int wave = threadIdx.x >> 5;
    const int rowBase = (blockIdx.x * 8 + wave) * 16;
    const int mrow = lane & 15;
    int row = rowBase + mrow;
    if (row >= nrows) row = nrows - 1;        // full-EXEC clamp (WMMA needs EXEC all-1)
    const float* zrow = z + (long)row * GMM_F;
    // A-layout K offset: low lanes own K{0..7,16..23}, high lanes K{8..15,24..31}
    const int off = (lane < 16) ? 0 : 8;

    v8f accz[4], accs[4];                     // independent z / z^2 chains
    #pragma unroll
    for (int t = 0; t < 4; ++t) {
        accz[t] = (v8f){0.f,0.f,0.f,0.f,0.f,0.f,0.f,0.f};
        accs[t] = (v8f){0.f,0.f,0.f,0.f,0.f,0.f,0.f,0.f};
    }

    for (int g = 0; g < 8; ++g) {
        if (g + 1 < 8)
            __builtin_prefetch((const void*)(zrow + (g + 1) * 32 + off), 0, 1);

        // 16 floats per lane: f = g*32 + off + {0..7}  and  + 16 + {0..7}
        const float4* zp = (const float4*)(zrow + g * 32 + off);
        float4 za = zp[0];
        float4 zb = zp[1];
        const float4* zq = (const float4*)(zrow + g * 32 + off + 16);
        float4 zc = zq[0];
        float4 zd = zq[1];
        float zv[16] = { za.x, za.y, za.z, za.w,  zb.x, zb.y, zb.z, zb.w,
                         zc.x, zc.y, zc.z, zc.w,  zd.x, zd.y, zd.z, zd.w };

        v16bf zh, zl, sh, sl;   // hi/lo splits of z and z^2
        #pragma unroll
        for (int e = 0; e < 16; ++e) {
            const float v  = zv[e];
            const __bf16 h = (__bf16)v;
            zh[e] = h;
            zl[e] = (__bf16)(v - (float)h);
            const float s   = v * v;
            const __bf16 hs = (__bf16)s;
            sh[e] = hs;
            sl[e] = (__bf16)(s - (float)hs);
        }

        #pragma unroll
        for (int t = 0; t < 4; ++t) {
            const int base1 = ((t * 16 + g)     * 32 + lane) * 16;  // w1 chunk
            const int base0 = ((t * 16 + 8 + g) * 32 + lane) * 16;  // w0 chunk
            v16bf b1h = *(const v16bf*)(Whi + base1);
            v16bf b1l = *(const v16bf*)(Wlo + base1);
            v16bf b0h = *(const v16bf*)(Whi + base0);
            v16bf b0l = *(const v16bf*)(Wlo + base0);

            // interleave the two chains: adjacent WMMAs are independent
            accz[t] = __builtin_amdgcn_wmma_f32_16x16x32_bf16(
                false, zh, false, b1h, (short)0, accz[t], false, false);
            accs[t] = __builtin_amdgcn_wmma_f32_16x16x32_bf16(
                false, sh, false, b0h, (short)0, accs[t], false, false);
            accz[t] = __builtin_amdgcn_wmma_f32_16x16x32_bf16(
                false, zl, false, b1h, (short)0, accz[t], false, false);
            accs[t] = __builtin_amdgcn_wmma_f32_16x16x32_bf16(
                false, sl, false, b0h, (short)0, accs[t], false, false);
            accz[t] = __builtin_amdgcn_wmma_f32_16x16x32_bf16(
                false, zh, false, b1l, (short)0, accz[t], false, false);
            accs[t] = __builtin_amdgcn_wmma_f32_16x16x32_bf16(
                false, sh, false, b0l, (short)0, accs[t], false, false);
        }
    }

    // ---- epilogue: lp = -0.5*(S + bias[c]); per-row logsumexp over 64 classes
    __shared__ float sm[8][16][64];
    float bc[4];
    #pragma unroll
    for (int t = 0; t < 4; ++t) bc[t] = bias[t * 16 + (lane & 15)];

    // C/D layout: VGPR r -> M = r (lanes 0-15) or 8+r (lanes 16-31), N = lane&15
    #pragma unroll
    for (int t = 0; t < 4; ++t) {
        #pragma unroll
        for (int r = 0; r < 8; ++r) {
            const int M = (lane < 16) ? r : (8 + r);
            sm[wave][M][t * 16 + (lane & 15)] =
                -0.5f * (accz[t][r] + accs[t][r] + bc[t]);
        }
    }
    __syncthreads();

    if (lane < 16) {
        const int orow = rowBase + lane;
        if (orow < nrows) {
            const float* p = &sm[wave][lane][0];
            float mx = -3.402823466e38f;
            for (int c = 0; c < 64; ++c) mx = fmaxf(mx, p[c]);
            float ssum = 0.0f;
            for (int c = 0; c < 64; ++c) ssum += expf(p[c] - mx);
            out[orow] = mx + logf(ssum) - LOG_C;
        }
    }
}

extern "C" void kernel_launch(void* const* d_in, const int* in_sizes, int n_in,
                              void* d_out, int out_size, void* d_ws, size_t ws_size,
                              hipStream_t stream) {
    const float* z        = (const float*)d_in[0];   // [B, 256]
    const float* means    = (const float*)d_in[1];   // [64, 256]
    const float* log_stds = (const float*)d_in[2];   // [64, 256]
    float* out            = (float*)d_out;           // [B]

    const int nrows = in_sizes[0] / GMM_F;           // B = 16384

    // workspace carve: Whi (64KB) | Wlo (64KB) | bias (256B)  -> ~128.25 KB
    __bf16* Whi = (__bf16*)d_ws;
    __bf16* Wlo = (__bf16*)((char*)d_ws + WPK_ELEMS * sizeof(__bf16));
    float*  bias = (float*)((char*)d_ws + 2 * WPK_ELEMS * sizeof(__bf16));

    gmm_prep_kernel<<<GMM_C, GMM_F, 0, stream>>>(means, log_stds, Whi, Wlo, bias);

    const int rowsPerBlock = 8 * 16;                  // 8 waves x 16-row WMMA tiles
    const int grid = (nrows + rowsPerBlock - 1) / rowsPerBlock;  // 128
    gmm_main_kernel<<<grid, 256, 0, stream>>>(z, Whi, Wlo, bias, out, nrows);
}
